// QRNNREDC3D_59124519796892
// MI455X (gfx1250) — matrix operand order repaired and checked by reading, not compiled, not run on hardware
//
#include <hip/hip_runtime.h>
#include <math.h>

// ---------------------------------------------------------------------------
// QRNN 3-D U-Net forward for MI455X (gfx1250, wave32, WMMA).
// Convs: implicit-GEMM, one wave computes a (Copad x 16) output tile:
// the B fragment (input patch, the expensive gather) is built ONCE per
// K-chunk and reused by MBLK = Copad/16 v_wmma_f32_16x16x32_f16 ops.
// Activations channels-last [D][H][W][C] fp32; weights repacked to f16
// [Copad][Kpad], K ordered (tap, ic), deconv-flip applied at pack time.
// upsample2x folded into the conv coordinate map (ups flag).
// Workspace requirement: ~1.11 GB (layout in kernel_launch).
// ---------------------------------------------------------------------------

typedef __attribute__((ext_vector_type(16))) _Float16 v16h;
typedef __attribute__((ext_vector_type(8)))  _Float16 v8h;
typedef __attribute__((ext_vector_type(8)))  float    v8f;
typedef __attribute__((ext_vector_type(4)))  float    v4f;

// ---------------------------------------------------------------------------
// Weight repack: fp32 OIDHW (3x3x3, optional flip) -> f16 [Copad][Kpad],
// K ordered as k = tap*Cin + ic, tap = (kd*3+kh)*3+kw. Zero padded.
// ---------------------------------------------------------------------------
__global__ void pack_weights_f16(const float* __restrict__ w, _Float16* __restrict__ dst,
                                 int CoutReal, int Copad, int Cin, int Kpad, int Ktot, int flip) {
  int idx = blockIdx.x * 256 + threadIdx.x;
  if (idx >= Copad * Kpad) return;
  int oc = idx / Kpad;
  int k  = idx - oc * Kpad;
  float v = 0.f;
  if (oc < CoutReal && k < Ktot) {
    int t  = k / Cin;
    int ic = k - t * Cin;
    int kd = t / 9;
    int r9 = t - kd * 9;
    int kh = r9 / 3;
    int kw = r9 - kh * 3;
    if (flip) { kd = 2 - kd; kh = 2 - kh; kw = 2 - kw; }
    v = w[(((size_t)oc * Cin + ic) * 3 + kd) * 9 + kh * 3 + kw];
  }
  dst[idx] = (_Float16)v;
}

// ---------------------------------------------------------------------------
// Implicit-GEMM conv3d (3x3x3, SAME pad, optional stride-2 or fused 2x
// upsample of the input) via WMMA f16 -> f32.
// One wave computes MBLK*16 (oc) x 16 (W-positions); B gathered once/chunk.
//   grid.x = D * Hout * (Wout/16), block = 32.
// in:   [Din][Hin][Win][Cin]  fp32 channels-last   (Cin in {16,32,64})
// wp:   [MBLK*16][Kpad]       f16 packed weights
// gout: [Din][Hout][Wout][MBLK*16] fp32 gates (bias added)
// ---------------------------------------------------------------------------
template <int MBLK>
__global__ __launch_bounds__(32) void conv3d_wmma_f16(
    const float* __restrict__ in, const _Float16* __restrict__ wp,
    const float* __restrict__ bias, float* __restrict__ gout,
    int Cin, int lc, int Kpad,
    int Din, int Hin, int Win,
    int Hout, int Wout,
    int stride, int ups,
    int CoutReal) {
  const int Copad = MBLK * 16;
  const int lane = threadIdx.x;
  const int hv   = lane >> 4;        // lane half (0/1)
  const int n    = lane & 15;        // B column / A row within a 16-block

  const int wtiles = Wout >> 4;
  int tile = blockIdx.x;
  int wt = tile % wtiles; tile /= wtiles;
  int h  = tile % Hout;
  int d  = tile / Hout;
  const int w = (wt << 4) + n;

  // Per-M-block weight row pointers (row n of each 16-row block).
  const _Float16* wrow[MBLK];
#pragma unroll
  for (int mb = 0; mb < MBLK; ++mb)
    wrow[mb] = wp + (size_t)(mb * 16 + n) * Kpad;

  const int Hb = ups ? (Hin << 1) : Hin;   // bounds in (possibly upsampled) space
  const int Wb = ups ? (Win << 1) : Win;

  v8f acc[MBLK];
#pragma unroll
  for (int mb = 0; mb < MBLK; ++mb) acc[mb] = (v8f){};

  for (int kc = 0; kc < Kpad; kc += 32) {
    // ---- B fragment (shared by all M blocks): lane holds
    // K = kc + 16*hv + (0..15) = one tap x 16 contiguous channels.
    int kB  = kc + (hv << 4);
    int t   = kB >> lc;              // tap index
    int ic0 = kB & (Cin - 1);        // first channel of the run
    v16h bfrag = {};
    if (t < 27) {
      int kd = t / 9;
      int r9 = t - kd * 9;
      int kh = r9 / 3;
      int kw = r9 - kh * 3;
      int di = d + kd - 1;
      int hy = h * stride + kh - 1;
      int wx = w * stride + kw - 1;
      if ((unsigned)di < (unsigned)Din && (unsigned)hy < (unsigned)Hb &&
          (unsigned)wx < (unsigned)Wb) {
        if (ups) { hy >>= 1; wx >>= 1; }
        const float* p = in + ((size_t)(di * Hin + hy) * Win + wx) * Cin + ic0;
        v4f f0 = *(const v4f*)(p);
        v4f f1 = *(const v4f*)(p + 4);
        v4f f2 = *(const v4f*)(p + 8);
        v4f f3 = *(const v4f*)(p + 12);
#pragma unroll
        for (int i = 0; i < 4; ++i) {
          bfrag[i]      = (_Float16)f0[i];
          bfrag[4 + i]  = (_Float16)f1[i];
          bfrag[8 + i]  = (_Float16)f2[i];
          bfrag[12 + i] = (_Float16)f3[i];
        }
      }
    }

    // ---- MBLK A fragments + WMMAs sharing bfrag.
#pragma unroll
    for (int mb = 0; mb < MBLK; ++mb) {
      __builtin_prefetch(wrow[mb] + kc + 32, 0, 1);
      v8h a0 = *(const v8h*)(wrow[mb] + kc + 16 * hv);
      v8h a1 = *(const v8h*)(wrow[mb] + kc + 16 * hv + 8);
      v16h afrag;
#pragma unroll
      for (int i = 0; i < 8; ++i) { afrag[i] = a0[i]; afrag[8 + i] = a1[i]; }
      acc[mb] = __builtin_amdgcn_wmma_f32_16x16x32_f16(false, afrag, false, bfrag,
                                                       (short)0, acc[mb], false, false);
    }
  }

  // ---- C/D store: lane (n, hv) holds rows oc0..oc0+7 at column w.
  float* gbase = gout + ((size_t)(d * Hout + h) * Wout + w) * Copad;
#pragma unroll
  for (int mb = 0; mb < MBLK; ++mb) {
    const int oc0 = mb * 16 + 8 * hv;
    float br[8];
#pragma unroll
    for (int r = 0; r < 8; ++r) {
      int oc = oc0 + r;
      br[r] = (oc < CoutReal) ? bias[oc] : 0.f;
    }
    v4f lo = {acc[mb][0] + br[0], acc[mb][1] + br[1], acc[mb][2] + br[2], acc[mb][3] + br[3]};
    v4f hi = {acc[mb][4] + br[4], acc[mb][5] + br[5], acc[mb][6] + br[6], acc[mb][7] + br[7]};
    *(v4f*)(gbase + oc0)     = lo;
    *(v4f*)(gbase + oc0 + 4) = hi;
  }
}

// ---------------------------------------------------------------------------
// First layer: Cin=1, Cout=48, direct conv with LDS-staged weights.
// x: [31][256][256] fp32, g: [N0][48] gates.
// ---------------------------------------------------------------------------
__global__ void conv_fe_48(const float* __restrict__ x, const float* __restrict__ w,
                           const float* __restrict__ b, float* __restrict__ g) {
  __shared__ float sw[48 * 27];
  __shared__ float sb[48];
  for (int i = threadIdx.x; i < 48 * 27; i += blockDim.x) sw[i] = w[i];
  if (threadIdx.x < 48) sb[threadIdx.x] = b[threadIdx.x];
  __syncthreads();

  size_t idx = (size_t)blockIdx.x * blockDim.x + threadIdx.x;
  int d  = (int)(idx >> 16);
  int hw = (int)(idx & 65535);
  int h  = hw >> 8;
  int wc = hw & 255;

  float v[27];
#pragma unroll
  for (int kd = 0; kd < 3; ++kd)
#pragma unroll
    for (int kh = 0; kh < 3; ++kh)
#pragma unroll
      for (int kw = 0; kw < 3; ++kw) {
        int di = d + kd - 1, hy = h + kh - 1, wx = wc + kw - 1;
        bool ok = (unsigned)di < 31u && (unsigned)hy < 256u && (unsigned)wx < 256u;
        v[(kd * 3 + kh) * 3 + kw] = ok ? x[((size_t)di * 256 + hy) * 256 + wx] : 0.f;
      }

  float* gp = g + idx * 48;
  for (int oc = 0; oc < 48; ++oc) {
    float s = sb[oc];
#pragma unroll
    for (int t = 0; t < 27; ++t) s += sw[oc * 27 + t] * v[t];
    gp[oc] = s;
  }
}

// ---------------------------------------------------------------------------
// QRNN fo-pool: h_d = f_d * h_{d-1} + (1-f_d) * tanh(z_d), scan over D.
// gates: [D][plane][Gpad] (z at [c], f at [C+c]); out: [D][plane][C].
// ---------------------------------------------------------------------------
__global__ void qrnn_pool(const float* __restrict__ g, float* __restrict__ out,
                          int C, int Gpad, int plane, int Dd, int rev) {
  size_t tid = (size_t)blockIdx.x * blockDim.x + threadIdx.x;
  if (tid >= (size_t)plane * C) return;
  int c  = (int)(tid % C);
  int hw = (int)(tid / C);
  float h = 0.f;
  for (int i = 0; i < Dd; ++i) {
    int dc = rev ? (Dd - 1 - i) : i;
    size_t base = (size_t)dc * plane + hw;
    float z = tanhf(g[base * Gpad + c]);
    float f = 1.f / (1.f + __expf(-g[base * Gpad + C + c]));
    h = f * h + (1.f - f) * z;
    out[base * C + c] = h;
  }
}

// biQRNN: forward pool with f1 plus backward pool with f2, shared z.
// Optional residual added once (used for the final "+ x").
__global__ void biqrnn_pool(const float* __restrict__ g, float* __restrict__ out,
                            const float* __restrict__ resid,
                            int C, int Gpad, int plane, int Dd) {
  size_t tid = (size_t)blockIdx.x * blockDim.x + threadIdx.x;
  if (tid >= (size_t)plane * C) return;
  int c  = (int)(tid % C);
  int hw = (int)(tid / C);
  float h = 0.f;
  for (int d = 0; d < Dd; ++d) {
    size_t base = (size_t)d * plane + hw;
    float z = tanhf(g[base * Gpad + c]);
    float f = 1.f / (1.f + __expf(-g[base * Gpad + C + c]));
    h = f * h + (1.f - f) * z;
    out[base * C + c] = h;
  }
  h = 0.f;
  for (int d = Dd - 1; d >= 0; --d) {
    size_t base = (size_t)d * plane + hw;
    float z = tanhf(g[base * Gpad + c]);
    float f = 1.f / (1.f + __expf(-g[base * Gpad + 2 * C + c]));
    h = f * h + (1.f - f) * z;
    size_t oi = base * C + c;
    float r = resid ? resid[oi] : 0.f;
    out[oi] += h + r;
  }
}

__global__ void add2(const float* __restrict__ a, const float* __restrict__ b,
                     float* __restrict__ c, size_t n) {
  size_t i = (size_t)blockIdx.x * blockDim.x + threadIdx.x;
  if (i < n) c[i] = a[i] + b[i];
}

// ---------------------------------------------------------------------------
// Host-side orchestration.
// ---------------------------------------------------------------------------
extern "C" void kernel_launch(void* const* d_in, const int* in_sizes, int n_in,
                              void* d_out, int out_size, void* d_ws, size_t ws_size,
                              hipStream_t stream) {
  (void)in_sizes; (void)n_in; (void)out_size; (void)ws_size;

  const float* x = (const float*)d_in[0];
  const float* Wt[12];
  const float* Bs[12];
  for (int j = 0; j < 12; ++j) {
    Wt[j] = (const float*)d_in[1 + 2 * j];
    Bs[j] = (const float*)d_in[2 + 2 * j];
  }
  enum { L_fe = 0, L_e0, L_e1, L_e2, L_e3, L_e4, L_d0, L_d1, L_d2, L_d3, L_d4, L_rec };

  const size_t N0 = 31ull * 256 * 256;   // 2,031,616
  const size_t N1 = 31ull * 128 * 128;   //   507,904
  const size_t N2 = 31ull * 64 * 64;     //   126,976

  // Workspace arena (floats). Total = 112*N0 + 64*N1 + 128*N2 ~= 276.3M floats
  // (~1.105 GB) + ~1.7 MB packed f16 weights.
  float* ws = (float*)d_ws;
  size_t off = 0;
  float* feb = ws + off; off += 16 * N0;   // fe   (16ch @ full)
  float* e0b = ws + off; off += 16 * N0;   // e0   (16ch @ full)
  float* e1b = ws + off; off += 32 * N1;   // e1   (32ch @ half)
  float* e2b = ws + off; off += 32 * N1;   // e2   (32ch @ half)
  float* e3b = ws + off; off += 64 * N2;   // e3   (64ch @ quarter)
  float* e4b = ws + off; off += 64 * N2;   // e4   (64ch @ quarter)
  float* G   = ws + off; off += 48 * N0;   // gates scratch (max 48ch @ full)
  float* T1  = ws + off; off += 16 * N0;   // decoder activations
  float* T2  = ws + off; off += 16 * N0;   // add results
  _Float16* wpk = (_Float16*)(ws + off);

  // WMMA conv layer table: {d_in layer, CoutReal, Copad, Cin, flip}
  struct LP { int li, CoutReal, Copad, Cin, flip; };
  const LP lps[11] = {
      {L_e0,  32,  32, 16, 0}, {L_e1,  64,  64, 16, 0}, {L_e2,  64,  64, 32, 0},
      {L_e3, 128, 128, 32, 0}, {L_e4, 128, 128, 64, 0}, {L_d0, 128, 128, 64, 1},
      {L_d1,  64,  64, 64, 0}, {L_d2,  64,  64, 32, 1}, {L_d3,  32,  32, 32, 0},
      {L_d4,  32,  32, 16, 1}, {L_rec,  3,  16, 16, 1}};
  size_t wo[11];
  {
    size_t a = 0;
    for (int i = 0; i < 11; ++i) {
      wo[i] = a;
      int Kpad = ((27 * lps[i].Cin) + 31) & ~31;
      a += (size_t)lps[i].Copad * Kpad;
    }
    for (int i = 0; i < 11; ++i) {
      int Ktot = 27 * lps[i].Cin;
      int Kpad = (Ktot + 31) & ~31;
      int tot  = lps[i].Copad * Kpad;
      pack_weights_f16<<<(tot + 255) / 256, 256, 0, stream>>>(
          Wt[lps[i].li], wpk + wo[i], lps[i].CoutReal, lps[i].Copad,
          lps[i].Cin, Kpad, Ktot, lps[i].flip);
    }
  }

  auto conv = [&](const float* in, int li, float* g, int Hin, int Win,
                  int Hout, int Wout, int stride, int ups) {
    const LP& L = lps[li];
    int Kpad = ((27 * L.Cin) + 31) & ~31;
    int lc = (L.Cin == 16) ? 4 : (L.Cin == 32) ? 5 : 6;
    dim3 grid((unsigned)(31 * Hout * (Wout / 16)));
    switch (L.Copad) {
      case 16:
        conv3d_wmma_f16<1><<<grid, 32, 0, stream>>>(in, wpk + wo[li], Bs[L.li], g,
            L.Cin, lc, Kpad, 31, Hin, Win, Hout, Wout, stride, ups, L.CoutReal);
        break;
      case 32:
        conv3d_wmma_f16<2><<<grid, 32, 0, stream>>>(in, wpk + wo[li], Bs[L.li], g,
            L.Cin, lc, Kpad, 31, Hin, Win, Hout, Wout, stride, ups, L.CoutReal);
        break;
      case 64:
        conv3d_wmma_f16<4><<<grid, 32, 0, stream>>>(in, wpk + wo[li], Bs[L.li], g,
            L.Cin, lc, Kpad, 31, Hin, Win, Hout, Wout, stride, ups, L.CoutReal);
        break;
      default:
        conv3d_wmma_f16<8><<<grid, 32, 0, stream>>>(in, wpk + wo[li], Bs[L.li], g,
            L.Cin, lc, Kpad, 31, Hin, Win, Hout, Wout, stride, ups, L.CoutReal);
        break;
    }
  };
  auto qrnn = [&](const float* g, float* out, int C, int Gpad, int plane, int rev) {
    size_t n = (size_t)plane * C;
    qrnn_pool<<<(unsigned)((n + 255) / 256), 256, 0, stream>>>(g, out, C, Gpad, plane, 31, rev);
  };
  auto addk = [&](const float* a, const float* b, float* c, size_t n) {
    add2<<<(unsigned)((n + 255) / 256), 256, 0, stream>>>(a, b, c, n);
  };

  // ---- encoder --------------------------------------------------------------
  conv_fe_48<<<(unsigned)(N0 / 256), 256, 0, stream>>>(x, Wt[L_fe], Bs[L_fe], G);
  biqrnn_pool<<<(unsigned)((65536ull * 16 + 255) / 256), 256, 0, stream>>>(
      G, feb, nullptr, 16, 48, 65536, 31);

  conv(feb, 0, G, 256, 256, 256, 256, 1, 0);           // e0 gates
  qrnn(G, e0b, 16, 32, 65536, 0);

  conv(e0b, 1, G, 256, 256, 128, 128, 2, 0);           // e1 gates (stride 2)
  qrnn(G, e1b, 32, 64, 16384, 1);

  conv(e1b, 2, G, 128, 128, 128, 128, 1, 0);           // e2 gates
  qrnn(G, e2b, 32, 64, 16384, 0);

  conv(e2b, 3, G, 128, 128, 64, 64, 2, 0);             // e3 gates (stride 2)
  qrnn(G, e3b, 64, 128, 4096, 1);

  conv(e3b, 4, G, 64, 64, 64, 64, 1, 0);               // e4 gates
  qrnn(G, e4b, 64, 128, 4096, 0);

  // ---- decoder --------------------------------------------------------------
  conv(e4b, 5, G, 64, 64, 64, 64, 1, 0);               // d0 gates (deconv)
  qrnn(G, T1, 64, 128, 4096, 1);

  addk(T1, e3b, T2, 64 * N2);                          // d0 + e3
  conv(T2, 6, G, 64, 64, 128, 128, 1, 1);              // d1 gates (fused upsample)
  qrnn(G, T1, 32, 64, 16384, 0);

  addk(T1, e2b, T2, 32 * N1);                          // d1 + e2
  conv(T2, 7, G, 128, 128, 128, 128, 1, 0);            // d2 gates (deconv)
  qrnn(G, T1, 32, 64, 16384, 1);

  addk(T1, e1b, T2, 32 * N1);                          // d2 + e1
  conv(T2, 8, G, 128, 128, 256, 256, 1, 1);            // d3 gates (fused upsample)
  qrnn(G, T1, 16, 32, 65536, 0);

  addk(T1, e0b, T2, 16 * N0);                          // d3 + e0
  conv(T2, 9, G, 256, 256, 256, 256, 1, 0);            // d4 gates (deconv)
  qrnn(G, T1, 16, 32, 65536, 1);

  addk(T1, feb, T2, 16 * N0);                          // d4 + fe
  conv(T2, 10, G, 256, 256, 256, 256, 1, 0);           // rec gates (deconv, Cout=3 -> 16)
  biqrnn_pool<<<(unsigned)((65536ull + 255) / 256), 256, 0, stream>>>(
      G, (float*)d_out, x, 1, 16, 65536, 31);          // biqrnn + residual x
}
